// MP_Layer_16484084482471
// MI455X (gfx1250) — compile-verified, matrix operand order checked by reading, not run on hardware
//
#include <hip/hip_runtime.h>

#define BNEPS 1e-3f

typedef __attribute__((ext_vector_type(16))) __bf16 v16bf;
typedef __attribute__((ext_vector_type(8)))  float  v8f;

union BF16Frag { v16bf v; unsigned short u[16]; };

__device__ __forceinline__ unsigned short f2bf(float f) {
  unsigned int u = __float_as_uint(f);
  u += 0x7FFFu + ((u >> 16) & 1u);   // round-to-nearest-even to bf16
  return (unsigned short)(u >> 16);
}

// Load one 16x16x32 WMMA operand fragment (A row / B column) from a run of
// bf16 values starting at p: lane holds K = {0..7} and {16..23} relative to p.
__device__ __forceinline__ v16bf load_frag(const unsigned short* p) {
  BF16Frag f;
#pragma unroll
  for (int t = 0; t < 8; ++t) { f.u[t] = p[t]; f.u[8 + t] = p[16 + t]; }
  return f.v;
}

__device__ __forceinline__ v8f wmma_bf16(v16bf a, v16bf b, v8f c) {
  return __builtin_amdgcn_wmma_f32_16x16x32_bf16(false, a, false, b, (short)0, c,
                                                 false, false);
}

// ---------------------------------------------------------------------------
// Weight prep: W (K x N row-major f32) -> Wt (N x K row-major bf16)
// ---------------------------------------------------------------------------
__global__ void convert_wt(const float* __restrict__ W,
                           unsigned short* __restrict__ Wt, int K, int N) {
  int idx = blockIdx.x * blockDim.x + threadIdx.x;
  if (idx >= K * N) return;
  int k = idx / N, n = idx - k * N;
  Wt[n * K + k] = f2bf(W[idx]);
}

// ---------------------------------------------------------------------------
// Edge kernel: one block per (batch b, target node i). Computes
//   new_edges = relu(relu([n_j, n_i, e] W1 + b1) W2 + b2) * mask      (adj MLP)
//   messages  = relu(relu([n_j, n_i, ne] W1 + b1) W2 + b2) * mask     (msg MLP)
//   agg[b,i,:] = sum_j messages                                       (in LDS)
//   out_edges = BN(new_edges)
// ---------------------------------------------------------------------------
__global__ __launch_bounds__(256) void edge_kernel(
    const float* __restrict__ nodes, const float* __restrict__ edges,
    const float* __restrict__ mask,
    const unsigned short* __restrict__ aW1t, const float* __restrict__ aB1,
    const unsigned short* __restrict__ aW2t, const float* __restrict__ aB2,
    const unsigned short* __restrict__ mW1t, const float* __restrict__ mB1,
    const unsigned short* __restrict__ mW2t, const float* __restrict__ mB2,
    const float* __restrict__ bnG, const float* __restrict__ bnB,
    const float* __restrict__ bnM, const float* __restrict__ bnV,
    float* __restrict__ outE, float* __restrict__ agg) {
  __shared__ __align__(16) unsigned char smem[16384 * 3 + 256 + 256];
  unsigned short* sNodes = (unsigned short*)smem;             // 64x128 bf16
  unsigned short* sEdge  = (unsigned short*)(smem + 16384);   // 64x128 bf16
  unsigned short* sH     = (unsigned short*)(smem + 32768);   // 64x128 bf16
  unsigned short* sRowI  = (unsigned short*)(smem + 49152);   // 128 bf16
  float*          sMask  = (float*)(smem + 49408);            // 64 f32
  float*          sMsg   = (float*)smem;  // 64x128 f32, aliases sNodes+sEdge
                                          // (only written after both are dead)

  const int tid  = threadIdx.x;
  const int lane = tid & 31;
  const int wave = tid >> 5;
  const int b = blockIdx.x >> 6;
  const int i = blockIdx.x & 63;

  const float* nodesB = nodes + b * (64 * 128);
  const float* edgesT = edges + (b * 4096 + i * 64) * 128;

  for (int idx = tid; idx < 64 * 128; idx += 256) {
    sNodes[idx] = f2bf(nodesB[idx]);   // state_i rows: nodes[j]
    sEdge[idx]  = f2bf(edgesT[idx]);   // edge features for rows i*64..i*64+63
  }
  if (tid < 128) sRowI[tid] = f2bf(nodesB[i * 128 + tid]);  // state_j = nodes[i]
  if (tid < 64)  sMask[tid] = mask[b * 4096 + i * 64 + tid];
  __syncthreads();

  const int lm   = lane & 15;
  const int kh   = ((lane >> 4) & 1) * 8;  // K sub-offset per lane half
  const int rsel = ((lane >> 4) & 1) * 8;  // C/D row offset per lane half
  const v8f vzero = {0.f, 0.f, 0.f, 0.f, 0.f, 0.f, 0.f, 0.f};

  for (int pass = 0; pass < 2; ++pass) {   // 0 = adj MLP, 1 = msg MLP
    const unsigned short* W1t = pass ? mW1t : aW1t;
    const unsigned short* W2t = pass ? mW2t : aW2t;
    const float* B1 = pass ? mB1 : aB1;
    const float* B2 = pass ? mB2 : aB2;

    v8f acc2[4];
#pragma unroll
    for (int q = 0; q < 4; ++q) acc2[q] = vzero;

    for (int half = 0; half < 2; ++half) {
      // layer 1: (64x384) @ (384 x 128-half) -> sH
      for (int q = 0; q < 4; ++q) {
        int T = wave * 4 + q;
        int m  = (T >> 3) * 16;
        int nl = (T & 7) * 16;
        int col = half * 128 + nl + lm;
        int row = m + lm;
        v8f acc = vzero;
#pragma unroll
        for (int kk = 0; kk < 12; ++kk) {
          int k0 = kk * 32 + kh;
          const unsigned short* ap =
              (k0 < 128) ? (sNodes + row * 128 + k0)
            : (k0 < 256) ? (sRowI + (k0 - 128))
                         : (sEdge + row * 128 + (k0 - 256));
          acc = wmma_bf16(load_frag(ap), load_frag(W1t + col * 384 + k0), acc);
        }
        float bias = B1[col];
        int rb = m + rsel;
#pragma unroll
        for (int r = 0; r < 8; ++r) {
          float v = fmaxf(acc[r] + bias, 0.f);
          sH[(rb + r) * 128 + nl + lm] = f2bf(v);
        }
      }
      __syncthreads();
      // layer 2 partial accumulate over K-range [half*128, half*128+128)
      for (int q = 0; q < 4; ++q) {
        int T = wave * 4 + q;
        int m = (T >> 3) * 16;
        int n = (T & 7) * 16;
        v8f acc = acc2[q];
#pragma unroll
        for (int kk = 0; kk < 4; ++kk) {
          int k0 = kk * 32 + kh;
          acc = wmma_bf16(load_frag(sH + (m + lm) * 128 + k0),
                          load_frag(W2t + (n + lm) * 256 + half * 128 + k0), acc);
        }
        acc2[q] = acc;
      }
      __syncthreads();
    }

    if (pass == 0) {
      // new_edges: BN'd copy to global, raw bf16 into sEdge for the msg MLP
      for (int q = 0; q < 4; ++q) {
        int T = wave * 4 + q;
        int m = (T >> 3) * 16;
        int n = (T & 7) * 16;
        int col = n + lm;
        int rb = m + rsel;
        float bias  = B2[col];
        float scale = bnG[col] * rsqrtf(bnV[col] + BNEPS);
        float shift = bnB[col] - bnM[col] * scale;
        v8f acc = acc2[q];
#pragma unroll
        for (int r = 0; r < 8; ++r) {
          int row = rb + r;
          float v = fmaxf(acc[r] + bias, 0.f) * sMask[row];
          outE[((b * 4096) + i * 64 + row) * 128 + col] = v * scale + shift;
          sEdge[row * 128 + col] = f2bf(v);
        }
      }
      __syncthreads();
    } else {
      // messages -> f32 LDS buffer (aliases dead sNodes/sEdge regions)
      for (int q = 0; q < 4; ++q) {
        int T = wave * 4 + q;
        int m = (T >> 3) * 16;
        int n = (T & 7) * 16;
        int col = n + lm;
        int rb = m + rsel;
        float bias = B2[col];
        v8f acc = acc2[q];
#pragma unroll
        for (int r = 0; r < 8; ++r) {
          int row = rb + r;
          sMsg[row * 128 + col] = fmaxf(acc[r] + bias, 0.f) * sMask[row];
        }
      }
      __syncthreads();
    }
  }

  // deterministic column reduction of the 64 messages -> agg[b,i,:]
  if (tid < 128) {
    float s = 0.f;
#pragma unroll 4
    for (int r = 0; r < 64; ++r) s += sMsg[r * 128 + tid];
    agg[(b * 64 + i) * 128 + tid] = s;
  }
}

// ---------------------------------------------------------------------------
// Update kernel: one block per batch. state = relu(relu([nodes,agg] W1+b1) W2+b2)
// ---------------------------------------------------------------------------
__global__ __launch_bounds__(256) void update_kernel(
    const float* __restrict__ nodes, const float* __restrict__ aggIn,
    const unsigned short* __restrict__ W1t, const float* __restrict__ B1,
    const unsigned short* __restrict__ W2t, const float* __restrict__ B2,
    float* __restrict__ stateOut) {
  __shared__ __align__(16) unsigned char smem[16384 * 3];
  unsigned short* sN = (unsigned short*)smem;
  unsigned short* sG = (unsigned short*)(smem + 16384);
  unsigned short* sH = (unsigned short*)(smem + 32768);

  const int tid  = threadIdx.x;
  const int lane = tid & 31;
  const int wave = tid >> 5;
  const int b = blockIdx.x;
  const int lm   = lane & 15;
  const int kh   = ((lane >> 4) & 1) * 8;
  const int rsel = ((lane >> 4) & 1) * 8;

  const float* nodesB = nodes + b * (64 * 128);
  const float* aggB   = aggIn + b * (64 * 128);
  for (int idx = tid; idx < 64 * 128; idx += 256) {
    sN[idx] = f2bf(nodesB[idx]);
    sG[idx] = f2bf(aggB[idx]);
  }
  __syncthreads();

  const v8f vzero = {0.f, 0.f, 0.f, 0.f, 0.f, 0.f, 0.f, 0.f};
  v8f acc2[4];
#pragma unroll
  for (int q = 0; q < 4; ++q) acc2[q] = vzero;

  for (int half = 0; half < 2; ++half) {
    for (int q = 0; q < 4; ++q) {
      int T = wave * 4 + q;
      int m  = (T >> 3) * 16;
      int nl = (T & 7) * 16;
      int col = half * 128 + nl + lm;
      int row = m + lm;
      v8f acc = vzero;
#pragma unroll
      for (int kk = 0; kk < 8; ++kk) {
        int k0 = kk * 32 + kh;
        const unsigned short* ap =
            (k0 < 128) ? (sN + row * 128 + k0) : (sG + row * 128 + (k0 - 128));
        acc = wmma_bf16(load_frag(ap), load_frag(W1t + col * 256 + k0), acc);
      }
      float bias = B1[col];
      int rb = m + rsel;
#pragma unroll
      for (int r = 0; r < 8; ++r) {
        float v = fmaxf(acc[r] + bias, 0.f);
        sH[(rb + r) * 128 + nl + lm] = f2bf(v);
      }
    }
    __syncthreads();
    for (int q = 0; q < 4; ++q) {
      int T = wave * 4 + q;
      int m = (T >> 3) * 16;
      int n = (T & 7) * 16;
      v8f acc = acc2[q];
#pragma unroll
      for (int kk = 0; kk < 4; ++kk) {
        int k0 = kk * 32 + kh;
        acc = wmma_bf16(load_frag(sH + (m + lm) * 128 + k0),
                        load_frag(W2t + (n + lm) * 256 + half * 128 + k0), acc);
      }
      acc2[q] = acc;
    }
    __syncthreads();
  }

  for (int q = 0; q < 4; ++q) {
    int T = wave * 4 + q;
    int m = (T >> 3) * 16;
    int n = (T & 7) * 16;
    int col = n + lm;
    int rb = m + rsel;
    float bias = B2[col];
    v8f acc = acc2[q];
#pragma unroll
    for (int r = 0; r < 8; ++r) {
      int row = rb + r;
      stateOut[(b * 64 + row) * 128 + col] = fmaxf(acc[r] + bias, 0.f);
    }
  }
}

// ---------------------------------------------------------------------------
// Final: state1 * sigmoid(state2), then both node batchnorms.
// ---------------------------------------------------------------------------
__global__ __launch_bounds__(256) void final_kernel(
    const float* __restrict__ s1, const float* __restrict__ s2,
    const float* __restrict__ g1, const float* __restrict__ be1,
    const float* __restrict__ mm1, const float* __restrict__ vv1,
    const float* __restrict__ g2, const float* __restrict__ be2,
    const float* __restrict__ mm2, const float* __restrict__ vv2,
    float* __restrict__ out0, float* __restrict__ out2) {
  int idx = blockIdx.x * 256 + threadIdx.x;
  if (idx >= 16 * 64 * 128) return;
  int c = idx & 127;
  float a = s1[idx], bb = s2[idx];
  float gated = a / (1.f + __expf(-bb));
  out0[idx] = (gated - mm1[c]) * rsqrtf(vv1[c] + BNEPS) * g1[c] + be1[c];
  out2[idx] = (bb - mm2[c]) * rsqrtf(vv2[c] + BNEPS) * g2[c] + be2[c];
}

// ---------------------------------------------------------------------------
extern "C" void kernel_launch(void* const* d_in, const int* in_sizes, int n_in,
                              void* d_out, int out_size, void* d_ws, size_t ws_size,
                              hipStream_t stream) {
  (void)in_sizes; (void)n_in; (void)out_size; (void)ws_size;

  const float* nodes1 = (const float*)d_in[0];
  const float* edges1 = (const float*)d_in[1];
  const float* nodes2 = (const float*)d_in[2];
  const float* edges2 = (const float*)d_in[3];
  const float* mask   = (const float*)d_in[4];
  // adj1, msg1, upd1, adj2, msg2, upd2: each (W1, b1, W2, b2)
  const float* adj1W1 = (const float*)d_in[5];
  const float* adj1b1 = (const float*)d_in[6];
  const float* adj1W2 = (const float*)d_in[7];
  const float* adj1b2 = (const float*)d_in[8];
  const float* msg1W1 = (const float*)d_in[9];
  const float* msg1b1 = (const float*)d_in[10];
  const float* msg1W2 = (const float*)d_in[11];
  const float* msg1b2 = (const float*)d_in[12];
  const float* upd1W1 = (const float*)d_in[13];
  const float* upd1b1 = (const float*)d_in[14];
  const float* upd1W2 = (const float*)d_in[15];
  const float* upd1b2 = (const float*)d_in[16];
  const float* adj2W1 = (const float*)d_in[17];
  const float* adj2b1 = (const float*)d_in[18];
  const float* adj2W2 = (const float*)d_in[19];
  const float* adj2b2 = (const float*)d_in[20];
  const float* msg2W1 = (const float*)d_in[21];
  const float* msg2b1 = (const float*)d_in[22];
  const float* msg2W2 = (const float*)d_in[23];
  const float* msg2b2 = (const float*)d_in[24];
  const float* upd2W1 = (const float*)d_in[25];
  const float* upd2b1 = (const float*)d_in[26];
  const float* upd2W2 = (const float*)d_in[27];
  const float* upd2b2 = (const float*)d_in[28];
  // bn_n1, bn_e1, bn_n2, bn_e2: each (gamma, beta, mean, var)
  const float* bnN1g = (const float*)d_in[29];
  const float* bnN1b = (const float*)d_in[30];
  const float* bnN1m = (const float*)d_in[31];
  const float* bnN1v = (const float*)d_in[32];
  const float* bnE1g = (const float*)d_in[33];
  const float* bnE1b = (const float*)d_in[34];
  const float* bnE1m = (const float*)d_in[35];
  const float* bnE1v = (const float*)d_in[36];
  const float* bnN2g = (const float*)d_in[37];
  const float* bnN2b = (const float*)d_in[38];
  const float* bnN2m = (const float*)d_in[39];
  const float* bnN2v = (const float*)d_in[40];
  const float* bnE2g = (const float*)d_in[41];
  const float* bnE2b = (const float*)d_in[42];
  const float* bnE2m = (const float*)d_in[43];
  const float* bnE2v = (const float*)d_in[44];

  // Output layout: bn(state1)[131072], bn(new_edges1)[8388608],
  //                bn(state2)[131072], bn(new_edges2)[8388608]
  float* out = (float*)d_out;
  float* out0 = out;
  float* out1 = out + 131072;
  float* out2 = out + 131072 + 8388608;
  float* out3 = out2 + 131072;

  // Workspace: transposed bf16 weights + agg + raw states
  unsigned short* wsu = (unsigned short*)d_ws;
  unsigned short* a1w1t = wsu;              // 256x384
  unsigned short* a1w2t = a1w1t + 98304;    // 128x256
  unsigned short* m1w1t = a1w2t + 32768;    // 256x384
  unsigned short* m1w2t = m1w1t + 98304;    // 128x256
  unsigned short* u1w1t = m1w2t + 32768;    // 256x256
  unsigned short* u1w2t = u1w1t + 65536;    // 128x256
  unsigned short* a2w1t = u1w2t + 32768;
  unsigned short* a2w2t = a2w1t + 98304;
  unsigned short* m2w1t = a2w2t + 32768;
  unsigned short* m2w2t = m2w1t + 98304;
  unsigned short* u2w1t = m2w2t + 32768;
  unsigned short* u2w2t = u2w1t + 65536;
  float* fws  = (float*)(u2w2t + 32768);
  float* agg1 = fws;
  float* agg2 = fws + 131072;
  float* st1  = fws + 262144;
  float* st2  = fws + 393216;

  auto conv = [&](const float* W, unsigned short* Wt, int K, int N) {
    convert_wt<<<(K * N + 255) / 256, 256, 0, stream>>>(W, Wt, K, N);
  };
  conv(adj1W1, a1w1t, 384, 256); conv(adj1W2, a1w2t, 256, 128);
  conv(msg1W1, m1w1t, 384, 256); conv(msg1W2, m1w2t, 256, 128);
  conv(upd1W1, u1w1t, 256, 256); conv(upd1W2, u1w2t, 256, 128);
  conv(adj2W1, a2w1t, 384, 256); conv(adj2W2, a2w2t, 256, 128);
  conv(msg2W1, m2w1t, 384, 256); conv(msg2W2, m2w2t, 256, 128);
  conv(upd2W1, u2w1t, 256, 256); conv(upd2W2, u2w2t, 256, 128);

  edge_kernel<<<1024, 256, 0, stream>>>(
      nodes1, edges1, mask, a1w1t, adj1b1, a1w2t, adj1b2,
      m1w1t, msg1b1, m1w2t, msg1b2,
      bnE1g, bnE1b, bnE1m, bnE1v, out1, agg1);
  edge_kernel<<<1024, 256, 0, stream>>>(
      nodes2, edges2, mask, a2w1t, adj2b1, a2w2t, adj2b2,
      m2w1t, msg2b1, m2w2t, msg2b2,
      bnE2g, bnE2b, bnE2m, bnE2v, out3, agg2);

  update_kernel<<<16, 256, 0, stream>>>(nodes1, agg1, u1w1t, upd1b1, u1w2t, upd1b2, st1);
  update_kernel<<<16, 256, 0, stream>>>(nodes2, agg2, u2w1t, upd2b1, u2w2t, upd2b2, st2);

  final_kernel<<<512, 256, 0, stream>>>(st1, st2,
                                        bnN1g, bnN1b, bnN1m, bnN1v,
                                        bnN2g, bnN2b, bnN2m, bnN2v,
                                        out0, out2);
}
